// SeqRNNTSU_3728031613722
// MI455X (gfx1250) — matrix operand order repaired and checked by reading, not compiled, not run on hardware
//
#include <hip/hip_runtime.h>

#define Bsz 256
#define Hsz 256
#define Ssz 512
#define BT  16
#define THREADS 512
#define HP  264   // padded row stride (ushorts) for bf16 LDS tiles
#define HF  260   // padded row stride (floats) for f32 h tile

typedef __attribute__((ext_vector_type(16))) __bf16 v16bf;
typedef __attribute__((ext_vector_type(8)))  float  v8f;

union Frag { v16bf v; uint4 q[2]; };

__device__ __forceinline__ unsigned short f2bf(float f) {
    union { float f; unsigned int u; } cv; cv.f = f;
    unsigned int u = cv.u + 0x7FFFu + ((cv.u >> 16) & 1u);   // RNE
    return (unsigned short)(u >> 16);
}

// Fast transcendentals: v_exp_f32 + v_rcp_f32 (single TRANS ops, no IEEE div sequence)
__device__ __forceinline__ float fast_sigmoid(float x) {
    return __builtin_amdgcn_rcpf(1.0f + __expf(-x));
}
__device__ __forceinline__ float fast_tanh(float x) {
    return 1.0f - 2.0f * __builtin_amdgcn_rcpf(__expf(2.0f * x) + 1.0f);
}

__global__ void cvt_weights_kernel(const float* __restrict__ wih,
                                   const float* __restrict__ whh,
                                   unsigned short* __restrict__ bih,
                                   unsigned short* __restrict__ bhh, int n) {
    int i = blockIdx.x * blockDim.x + threadIdx.x;
    if (i < n) { bih[i] = f2bf(wih[i]); bhh[i] = f2bf(whh[i]); }
}

#define WMMA_BF16(A, B, C) \
    __builtin_amdgcn_wmma_f32_16x16x32_bf16(false, (A), false, (B), (short)0, (C), false, false)

__launch_bounds__(THREADS)
__global__ void gru_scan_kernel(const float* __restrict__ tree,
                                const float* __restrict__ seq,
                                const int*   __restrict__ mask,
                                const float* __restrict__ b_ih,
                                const float* __restrict__ b_hh,
                                const unsigned short* __restrict__ wb_ih,
                                const unsigned short* __restrict__ wb_hh,
                                float* __restrict__ out) {
    __shared__ __align__(16) unsigned short xstage[4 * BT * HP]; // 4 staged timesteps, bf16
    __shared__ __align__(16) unsigned short h_bf[BT * HP];       // h as bf16 (WMMA A operand)
    __shared__ __align__(16) float          h_f32[BT * HF];      // h full precision
    __shared__ int lastidx[BT];

    const int tid = threadIdx.x;
    const int b0  = blockIdx.x * BT;
    const int w   = tid >> 5;      // wave id 0..15 (wave32)
    const int L   = tid & 31;      // lane
    const int hi  = L >> 4;        // half-wave select
    const int l15 = L & 15;

    // ---- prologue: last unmasked index per row ----
    if (tid < BT) {
        const int* mrow = mask + (size_t)(b0 + tid) * Ssz;
        int s = 0;
        for (int t = 0; t < Ssz; ++t) s += mrow[t];
        int li = s - 1; if (li < 0) li = 0;
        lastidx[tid] = li;
    }
    // ---- prologue: h <- tree_state ----
    for (int q = 0; q < (BT * Hsz) / THREADS; ++q) {
        int p  = tid + q * THREADS;       // p = i*256 + j
        int i  = p >> 8, j = p & 255;
        float v = tree[(size_t)b0 * Hsz + p];  // (b0+i)*H + j
        h_f32[i * HF + j] = v;
        h_bf [i * HP + j] = f2bf(v);
    }

    // per-lane constants: owned hidden column = w*16 + l15
    const int col = w * 16 + l15;
    const float br  = b_ih[col]       + b_hh[col];
    const float bz  = b_ih[256 + col] + b_hh[256 + col];
    const float bin = b_ih[512 + col];
    const float bhn = b_hh[512 + col];

    // B-fragment row base pointers (W stored (3H,H) row-major = (n,k) order),
    // pre-offset by the per-lane half-select (hi*8 K elements).
    const int hofs = hi * 8;
    const unsigned short* wr_i = wb_ih + (size_t)(      col) * Hsz + hofs;
    const unsigned short* wr_h = wb_hh + (size_t)(      col) * Hsz + hofs;
    const unsigned short* wz_i = wb_ih + (size_t)(256 + col) * Hsz + hofs;
    const unsigned short* wz_h = wb_hh + (size_t)(256 + col) * Hsz + hofs;
    const unsigned short* wn_i = wb_ih + (size_t)(512 + col) * Hsz + hofs;
    const unsigned short* wn_h = wb_hh + (size_t)(512 + col) * Hsz + hofs;

    __syncthreads();

    for (int t = 0; t < Ssz; ++t) {
        if ((t & 3) == 0) {
            // stage x for t..t+3: contiguous float4 along seq axis, convert to bf16
            for (int q = 0; q < (BT * Hsz) / THREADS; ++q) {
                int p = tid + q * THREADS;          // p = i*256 + h
                int i = p >> 8, hc = p & 255;
                const float4 xv = *(const float4*)(seq + ((size_t)b0 * Hsz + p) * Ssz + t);
                int base = i * HP + hc;
                xstage[0 * BT * HP + base] = f2bf(xv.x);
                xstage[1 * BT * HP + base] = f2bf(xv.y);
                xstage[2 * BT * HP + base] = f2bf(xv.z);
                xstage[3 * BT * HP + base] = f2bf(xv.w);
            }
            __syncthreads();
        }
        const unsigned short* xs = xstage + (t & 3) * (BT * HP);

        v8f accR = {}, accZ = {}, accIN = {}, accHN = {};
        const int arow = l15 * HP + hi * 8;
        // NOTE: unroll 1 keeps the weight loads inside the loop (addresses depend
        // on the live induction variable), preventing LICM hoist + VGPR spill.
        #pragma unroll 1
        for (int kk = 0; kk < 8; ++kk) {
            const int k0 = kk * 32;
            Frag ax, ah, fri, frh, fzi, fzh, fni, fnh;
            // A fragments (LDS): halves K = k0+hi*8+{0..7}, then +16
            ax.q[0] = *(const uint4*)(xs   + arow + k0);
            ax.q[1] = *(const uint4*)(xs   + arow + k0 + 16);
            ah.q[0] = *(const uint4*)(h_bf + arow + k0);
            ah.q[1] = *(const uint4*)(h_bf + arow + k0 + 16);
            // B fragments (global, L2-resident): halves K = k0+hi*8+{0..7}, then +16 (hi*8 pre-applied)
            fri.q[0] = *(const uint4*)(wr_i + k0); fri.q[1] = *(const uint4*)(wr_i + k0 + 16);
            frh.q[0] = *(const uint4*)(wr_h + k0); frh.q[1] = *(const uint4*)(wr_h + k0 + 16);
            fzi.q[0] = *(const uint4*)(wz_i + k0); fzi.q[1] = *(const uint4*)(wz_i + k0 + 16);
            fzh.q[0] = *(const uint4*)(wz_h + k0); fzh.q[1] = *(const uint4*)(wz_h + k0 + 16);
            fni.q[0] = *(const uint4*)(wn_i + k0); fni.q[1] = *(const uint4*)(wn_i + k0 + 16);
            fnh.q[0] = *(const uint4*)(wn_h + k0); fnh.q[1] = *(const uint4*)(wn_h + k0 + 16);

            accR  = WMMA_BF16(ax.v, fri.v, accR);
            accZ  = WMMA_BF16(ax.v, fzi.v, accZ);
            accIN = WMMA_BF16(ax.v, fni.v, accIN);
            accR  = WMMA_BF16(ah.v, frh.v, accR);
            accZ  = WMMA_BF16(ah.v, fzh.v, accZ);
            accHN = WMMA_BF16(ah.v, fnh.v, accHN);
        }
        __syncthreads();   // all waves done reading h_bf / xstage

        // gate math directly on accumulator fragments: VGPR g -> M = g + 8*hi, N = col
        const int m0 = hi * 8;
        #pragma unroll
        for (int g = 0; g < 8; ++g) {
            const int m = m0 + g;
            float r  = fast_sigmoid(accR[g] + br);
            float z  = fast_sigmoid(accZ[g] + bz);
            float nn = fast_tanh(accIN[g] + bin + r * (accHN[g] + bhn));
            float hold = h_f32[m * HF + col];
            float hnew = (1.0f - z) * nn + z * hold;
            h_f32[m * HF + col] = hnew;
            h_bf [m * HP + col] = f2bf(hnew);
            if (lastidx[m] == t) out[(size_t)(b0 + m) * Hsz + col] = hnew;
        }
        __syncthreads();   // h updates visible before next step's GEMM / staging
    }
}

extern "C" void kernel_launch(void* const* d_in, const int* in_sizes, int n_in,
                              void* d_out, int out_size, void* d_ws, size_t ws_size,
                              hipStream_t stream) {
    const float* tree  = (const float*)d_in[0];   // (B,H)
    const float* seq   = (const float*)d_in[1];   // (B,H,S)
    const int*   msk   = (const int*)  d_in[2];   // (B,S)
    const float* W_ih  = (const float*)d_in[3];   // (3H,H)
    const float* W_hh  = (const float*)d_in[4];   // (3H,H)
    const float* b_ih  = (const float*)d_in[5];   // (3H)
    const float* b_hh  = (const float*)d_in[6];   // (3H)
    float* out = (float*)d_out;                   // (B,H)

    const int wn = 3 * Hsz * Hsz;                 // 196608 per matrix
    unsigned short* wb_ih = (unsigned short*)d_ws;
    unsigned short* wb_hh = wb_ih + wn;

    cvt_weights_kernel<<<(wn + 255) / 256, 256, 0, stream>>>(W_ih, W_hh, wb_ih, wb_hh, wn);
    gru_scan_kernel<<<Bsz / BT, THREADS, 0, stream>>>(tree, seq, msk, b_ih, b_hh,
                                                      wb_ih, wb_hh, out);
}
